// pointer_net_33311766348198
// MI455X (gfx1250) — matrix operand order repaired and checked by reading, not compiled, not run on hardware
//
#include <hip/hip_runtime.h>

// ============================ CDNA5 WMMA types ============================
typedef __attribute__((ext_vector_type(16))) _Float16 v16h;
typedef __attribute__((ext_vector_type(8)))  _Float16 v8h;
typedef __attribute__((ext_vector_type(8)))  float    v8f;

// Global (addrspace(1)) views so laundered pointers still emit global_load_b128
// (FLAT loads would tick both LOADcnt and DScnt and couple against the LDS pipe).
typedef _Float16 __attribute__((address_space(1))) GF16;
typedef v16h     __attribute__((address_space(1))) GV16H;

union V8F  { v8f  v; float    f[8];  };
union V8H  { v8h  v; _Float16 h[8];  };

// Problem constants
#define NB   64      // batch
#define NS   128     // seq len
#define NE   128     // embed dim
#define NH   256     // hidden
#define NG   1024    // 4*H gates

#define NT_LSTM 512  // threads for persistent recurrent kernels (16 waves -> 4/SIMD)

#define LOG_EPS (-103.2789307f)   // logf(float(1e-45)) (smallest subnormal)

static __device__ __forceinline__ float sigf(float x){ return 1.0f/(1.0f+expf(-x)); }

// Prevent LICM from hoisting weight loads out of the time loop (which spills):
// make the base pointer opaque per iteration; loads then re-issue and hit L2.
#define LAUNDER_PTR(p) asm volatile("" : "+s"(p))

// A-fragment (16x32 f16) swizzled LDS layout. For element (m, k):
//   lanes 0-15 hold K sets {0-7,16-23}, lanes 16-31 hold {8-15,24-31} (ISA 7.12.2).
//   idx = ((m_tile*KT + k_tile)*32 + lane)*16 + j
static __device__ __forceinline__ int a_swz(int m, int k, int KT){
  int kt = k >> 5, ki = k & 31;
  int hi = (ki >> 3) & 1;
  int j  = (ki & 7) + ((ki >> 4) & 1) * 8;
  int lane = hi*16 + (m & 15);
  return (((m >> 4)*KT + kt)*32 + lane)*16 + j;
}
// B-fragment (32x16 f16) swizzle for weight row r (output), k (reduction)
static __device__ __forceinline__ int b_swz(int r, int k, int KT){
  int kt = k >> 5, ki = k & 31;
  int hi = (ki >> 3) & 1, j = (ki & 7) + ((ki >> 4) & 1) * 8;
  int lane = hi*16 + (r & 15);
  return (((r >> 4)*KT + kt)*32 + lane)*16 + j;
}

static __device__ __forceinline__ v8f wmma_f16(v16h a, v16h b, v8f c){
  return __builtin_amdgcn_wmma_f32_16x16x32_f16(false, a, false, b, (short)0, c, false, false);
}

// ============================ prep kernels ============================

__global__ void pn_embed(const float* __restrict__ x, const float* __restrict__ We,
                         _Float16* __restrict__ emb){
  int i = blockIdx.x*blockDim.x + threadIdx.x;     // < 64*128*128
  int e = i & 127, bt = i >> 7;
  float v = x[bt*2+0]*We[e*2+0] + x[bt*2+1]*We[e*2+1];
  emb[i] = (_Float16)v;
}

// WMMA-B swizzled f16 weights: cols [0,KH) from Wmain (RxKH), cols [KH,KH+KX) from Wx
__global__ void pn_swzB(_Float16* __restrict__ dst, const float* __restrict__ Wmain,
                        const float* __restrict__ Wx, int R, int KH, int KX){
  int K = KH + KX, KT = K >> 5;
  long total = (long)R * K;
  for (long i = blockIdx.x*(long)blockDim.x + threadIdx.x; i < total;
       i += (long)gridDim.x*blockDim.x){
    int r = (int)(i / K), k = (int)(i - (long)r*K);
    float val = (k < KH) ? Wmain[(long)r*KH + k] : Wx[(long)r*KX + (k - KH)];
    dst[b_swz(r, k, KT)] = (_Float16)val;
  }
}

__global__ void pn_bias(float* __restrict__ dst, const float* __restrict__ a,
                        const float* __restrict__ b, int n){
  int i = blockIdx.x*blockDim.x + threadIdx.x;
  if (i < n) dst[i] = a[i] + b[i];
}

// ============================ bidirectional LSTM ============================
// grid = 2 blocks (0: fwd, 1: bwd), 512 threads = 16 waves (4 waves/SIMD for VGPRs).
// Per step: g = [h ; x_t] @ Wcat^T (64 x 384 x 1024) via v_wmma_f32_16x16x32_f16.
// Each wave owns 4 (m_tile, n_tile) combos (same m_tile) across all 128 steps;
// c and h live in v8f accumulator-layout registers for the whole kernel.
__global__ void __launch_bounds__(NT_LSTM)
pn_bilstm(const _Float16* __restrict__ emb, const int* __restrict__ len_g,
          const _Float16* __restrict__ WcatF, const _Float16* __restrict__ WcatB,
          const float* __restrict__ biasF, const float* __restrict__ biasB,
          _Float16* __restrict__ encF, _Float16* __restrict__ encB,
          float* __restrict__ hT, float* __restrict__ cT){
  extern __shared__ char smem[];
  _Float16* Abuf  = (_Float16*)smem;            // 4 m_tiles * 12 k_tiles * 512 = 24576 halves
  float*    biasS = (float*)(smem + 49152);     // 1024 f32
  int*      lenS  = (int*)  (smem + 53248);     // 64
  const bool bwd = (blockIdx.x == 1);
  const _Float16* Wcat = bwd ? WcatB : WcatF;
  const float*    bias = bwd ? biasB : biasF;
  _Float16*       encO = bwd ? encB  : encF;

  const int tid = threadIdx.x, lane = tid & 31, wave = tid >> 5;
  // warm the L2 with the recurrent weights (global_prefetch_b8), 128B granularity
  for (size_t p = (size_t)tid*128; p < (size_t)NG*384*2; p += (size_t)NT_LSTM*128)
    __builtin_prefetch((const char*)Wcat + p, 0, 1);
  for (int i = tid; i < NG; i += NT_LSTM) biasS[i] = bias[i];
  if (tid < NB) lenS[tid] = len_g[tid];
  for (int i = tid; i < 4*12*512; i += NT_LSTM) Abuf[i] = (_Float16)0.0f;  // h0 = 0
  __syncthreads();

  // stage x_t (with in-length reversal for bwd) into A k-range [256,384)
  auto stage_x = [&](int t){
    for (int i = tid; i < NB*NE; i += NT_LSTM){
      int b = i >> 7, e = i & 127;
      int ts = t;
      if (bwd){ int L = lenS[b]; ts = (t < L) ? (L - 1 - t) : t; }
      Abuf[a_swz(b, 256 + e, 12)] = emb[(b*NS + ts)*NE + e];
    }
  };
  stage_x(0);

  V8F cst[4], hst[4];
  #pragma unroll
  for (int q = 0; q < 4; ++q)
    #pragma unroll
    for (int v = 0; v < 8; ++v){ cst[q].f[v] = 0.0f; hst[q].f[v] = 0.0f; }
  __syncthreads();

  const int mt  = (wave*4) >> 4;              // shared by all 4 combos of this wave
  const int nl  = lane & 15, hiL = lane >> 4;

  const GF16* Wt = (const GF16*)Wcat;
  for (int t = 0; t < NS; ++t){
    LAUNDER_PTR(Wt);   // force per-step weight loads (L2 hits), no cross-step hoist
    #pragma unroll
    for (int q = 0; q < 4; ++q){
      const int nt = (wave*4 + q) & 15;
      V8F acc[4];
      #pragma unroll
      for (int g = 0; g < 4; ++g)
        #pragma unroll
        for (int v = 0; v < 8; ++v) acc[g].f[v] = 0.0f;
      for (int kt = 0; kt < 12; ++kt){
        v16h a = *(const v16h*)&Abuf[((mt*12 + kt)*32 + lane)*16];
        #pragma unroll
        for (int g = 0; g < 4; ++g){
          v16h bfrag = *(const GV16H*)(Wt + (((g*16 + nt)*12 + kt)*32 + lane)*16);
          acc[g].v = wmma_f16(a, bfrag, acc[g].v);
        }
      }
      const int n = nt*16 + nl;
      float bi = biasS[n], bf = biasS[256+n], bg = biasS[512+n], bo = biasS[768+n];
      #pragma unroll
      for (int v = 0; v < 8; ++v){
        int m = mt*16 + v + 8*hiL;
        int L = lenS[m];
        float gi = sigf(acc[0].f[v] + bi);
        float gf = sigf(acc[1].f[v] + bf);
        float gg = tanhf(acc[2].f[v] + bg);
        float go = sigf(acc[3].f[v] + bo);
        float cn = gf*cst[q].f[v] + gi*gg;
        float hn = go*tanhf(cn);
        bool valid = (t < L);
        cst[q].f[v] = valid ? cn : cst[q].f[v];
        hst[q].f[v] = valid ? hn : hst[q].f[v];
        float hout = valid ? hn : 0.0f;
        int tout = t;
        if (bwd) tout = valid ? (L - 1 - t) : t;   // undo reversal on output
        encO[((size_t)m*NS + tout)*NH + n] = (_Float16)hout;
      }
    }
    __syncthreads();
    // publish new h into A k-range [0,256) and stage next x
    #pragma unroll
    for (int q = 0; q < 4; ++q){
      const int nt = (wave*4 + q) & 15;
      const int n = nt*16 + nl;
      #pragma unroll
      for (int v = 0; v < 8; ++v){
        int m = mt*16 + v + 8*hiL;
        Abuf[a_swz(m, n, 12)] = (_Float16)hst[q].f[v];
      }
    }
    if (t + 1 < NS) stage_x(t + 1);
    __syncthreads();
  }
  if (!bwd){ // decoder initial state = forward (hT, cT)
    #pragma unroll
    for (int q = 0; q < 4; ++q){
      const int nt = (wave*4 + q) & 15;
      const int n = nt*16 + nl;
      #pragma unroll
      for (int v = 0; v < 8; ++v){
        int m = mt*16 + v + 8*hiL;
        hT[m*NH + n] = hst[q].f[v];
        cT[m*NH + n] = cst[q].f[v];
      }
    }
  }
}

// enc = enc_f + enc_b (kept f16 for WMMA-A reuse and dec_in gather)
__global__ void pn_enc_combine(const _Float16* __restrict__ a, const _Float16* __restrict__ b,
                               _Float16* __restrict__ s){
  int i = blockIdx.x*blockDim.x + threadIdx.x;
  s[i] = (_Float16)((float)a[i] + (float)b[i]);
}

// enc_trans = enc @ W1^T : M=8192, N=256, K=256. One 16x16 tile per wave.
__global__ void __launch_bounds__(256)
pn_enc_trans(const _Float16* __restrict__ encS, const _Float16* __restrict__ W1s,
             float* __restrict__ encT){
  int lane = threadIdx.x & 31, wave = threadIdx.x >> 5;
  int tile = blockIdx.x*8 + wave;        // 8192 tiles
  int mt = tile >> 4, nt = tile & 15;
  int mrow = mt*16 + (lane & 15), hiL = lane >> 4;
  V8F acc;
  #pragma unroll
  for (int v = 0; v < 8; ++v) acc.f[v] = 0.0f;
  for (int kt = 0; kt < 8; ++kt){
    const _Float16* rp = encS + (size_t)mrow*NH + kt*32 + hiL*8;
    V8H lo, hi2; lo.v = *(const v8h*)rp; hi2.v = *(const v8h*)(rp + 16);
    v16h a;
    #pragma unroll
    for (int i = 0; i < 8; ++i){ a[i] = lo.h[i]; a[i+8] = hi2.h[i]; }
    v16h bfrag = *(const v16h*)&W1s[((nt*8 + kt)*32 + lane)*16];
    acc.v = wmma_f16(a, bfrag, acc.v);
  }
  int n = nt*16 + (lane & 15);
  #pragma unroll
  for (int v = 0; v < 8; ++v){
    int m = mt*16 + v + 8*hiL;
    encT[(size_t)m*NH + n] = acc.f[v];
  }
}

// mask_tensor output (as 0/1 floats)
__global__ void pn_mask(const int* __restrict__ len, float* __restrict__ out){
  int i = blockIdx.x*blockDim.x + threadIdx.x;   // 64*128*128
  int j = i & 127, r = (i >> 7) & 127, b = i >> 14;
  int L = len[b];
  out[i] = (r < L && j < L) ? 1.0f : 0.0f;
}

// ============================ pointer decoder ============================
// One persistent workgroup (512 threads = 16 waves): per step
// gate GEMM (K=512) -> q = h@W2^T -> attention -> masked log-softmax/argmax -> gather.
__global__ void __launch_bounds__(NT_LSTM)
pn_decoder(const _Float16* __restrict__ WcatD, const float* __restrict__ biasD,
           const _Float16* __restrict__ W2s, const float* __restrict__ hT,
           const float* __restrict__ cT, const _Float16* __restrict__ encS,
           const float* __restrict__ encT, const float* __restrict__ Vg,
           const int* __restrict__ len_g, float* __restrict__ out_ls,
           float* __restrict__ out_am){
  extern __shared__ char smem[];
  _Float16* Abuf  = (_Float16*)smem;             // 4*16*512 halves = 64KB: [h | dec_in]
  float*    qbuf  = (float*)(smem + 65536);      // 64*256 f32
  float*    ubuf  = (float*)(smem + 131072);     // 64*128 f32
  float*    biasS = (float*)(smem + 163840);     // 1024 f32
  float*    vS    = (float*)(smem + 167936);     // 256 f32
  int*      lenS  = (int*)  (smem + 168960);     // 64
  int*      amS   = (int*)  (smem + 169216);     // 64

  const int tid = threadIdx.x, lane = tid & 31, wave = tid >> 5;
  for (size_t p = (size_t)tid*128; p < (size_t)NG*512*2; p += (size_t)NT_LSTM*128)
    __builtin_prefetch((const char*)WcatD + p, 0, 1);
  for (int i = tid; i < NG; i += NT_LSTM) biasS[i] = biasD[i];
  for (int i = tid; i < NH; i += NT_LSTM) vS[i] = Vg[i];
  if (tid < NB) lenS[tid] = len_g[tid];
  for (int i = tid; i < 4*16*512; i += NT_LSTM) Abuf[i] = (_Float16)0.0f; // dec_in0 = 0
  __syncthreads();

  const int mt  = (wave*4) >> 4;
  const int nl  = lane & 15, hiL = lane >> 4;

  V8F cst[4], hst[4];
  #pragma unroll
  for (int q = 0; q < 4; ++q){
    const int nt = (wave*4 + q) & 15;
    const int n = nt*16 + nl;
    #pragma unroll
    for (int v = 0; v < 8; ++v){
      int m = mt*16 + v + 8*hiL;
      float hv = hT[m*NH + n];
      hst[q].f[v] = hv; cst[q].f[v] = cT[m*NH + n];
      Abuf[a_swz(m, n, 16)] = (_Float16)hv;
    }
  }
  __syncthreads();

  const GF16* WD  = (const GF16*)WcatD;
  const GF16* W2t = (const GF16*)W2s;
  for (int t = 0; t < NS; ++t){
    LAUNDER_PTR(WD);
    LAUNDER_PTR(W2t);
    // ---- 1) gates = [h ; dec_in] @ WcatD^T, LSTM cell ----
    #pragma unroll
    for (int q = 0; q < 4; ++q){
      const int nt = (wave*4 + q) & 15;
      V8F acc[4];
      #pragma unroll
      for (int g = 0; g < 4; ++g)
        #pragma unroll
        for (int v = 0; v < 8; ++v) acc[g].f[v] = 0.0f;
      for (int kt = 0; kt < 16; ++kt){
        v16h a = *(const v16h*)&Abuf[((mt*16 + kt)*32 + lane)*16];
        #pragma unroll
        for (int g = 0; g < 4; ++g){
          v16h bfrag = *(const GV16H*)(WD + (((g*16 + nt)*16 + kt)*32 + lane)*16);
          acc[g].v = wmma_f16(a, bfrag, acc[g].v);
        }
      }
      const int n = nt*16 + nl;
      float bi = biasS[n], bf = biasS[256+n], bg = biasS[512+n], bo = biasS[768+n];
      #pragma unroll
      for (int v = 0; v < 8; ++v){
        float gi = sigf(acc[0].f[v] + bi);
        float gf = sigf(acc[1].f[v] + bf);
        float gg = tanhf(acc[2].f[v] + bg);
        float go = sigf(acc[3].f[v] + bo);
        float cn = gf*cst[q].f[v] + gi*gg;
        cst[q].f[v] = cn;
        hst[q].f[v] = go*tanhf(cn);
      }
    }
    __syncthreads();
    // ---- 2) publish h (A k-range [0,256)) ----
    #pragma unroll
    for (int q = 0; q < 4; ++q){
      const int nt = (wave*4 + q) & 15;
      const int n = nt*16 + nl;
      #pragma unroll
      for (int v = 0; v < 8; ++v){
        int m = mt*16 + v + 8*hiL;
        Abuf[a_swz(m, n, 16)] = (_Float16)hst[q].f[v];
      }
    }
    __syncthreads();
    // ---- 3) q = h @ W2^T ----
    #pragma unroll
    for (int q = 0; q < 4; ++q){
      const int nt = (wave*4 + q) & 15;
      V8F acc;
      #pragma unroll
      for (int v = 0; v < 8; ++v) acc.f[v] = 0.0f;
      for (int kt = 0; kt < 8; ++kt){
        v16h a = *(const v16h*)&Abuf[((mt*16 + kt)*32 + lane)*16];
        v16h bfrag = *(const GV16H*)(W2t + ((nt*8 + kt)*32 + lane)*16);
        acc.v = wmma_f16(a, bfrag, acc.v);
      }
      const int n = nt*16 + nl;
      #pragma unroll
      for (int v = 0; v < 8; ++v){
        int m = mt*16 + v + 8*hiL;
        qbuf[m*NH + n] = acc.f[v];
      }
    }
    __syncthreads();
    // ---- 4) u[b,s] = v . tanh(enc_trans[b,s,:] + q[b,:]) ----
    for (int p = 0; p < 512; ++p){
      int pi = wave*512 + p;
      int b = pi >> 7, s = pi & 127;
      const float* ep = encT + ((size_t)(b*NS + s))*NH + lane*8;
      const float* qp = qbuf + b*NH + lane*8;
      float part = 0.0f;
      #pragma unroll
      for (int u2 = 0; u2 < 8; ++u2)
        part += vS[lane*8 + u2] * tanhf(ep[u2] + qp[u2]);
      #pragma unroll
      for (int off = 16; off > 0; off >>= 1)
        part += __shfl_xor(part, off, 32);
      if (lane == 0) ubuf[pi] = part;
    }
    __syncthreads();
    // ---- 5) masked log-softmax + argmax per row (4 rows per wave) ----
    for (int rq = 0; rq < 4; ++rq){
      int b = wave*4 + rq;
      int L = lenS[b];
      bool rowv = (t < L);
      float lg[4], mx = -1e30f;
      #pragma unroll
      for (int j2 = 0; j2 < 4; ++j2){
        int s = lane + 32*j2;
        bool msk = rowv && (s < L);
        float lv = ubuf[b*NS + s] + (msk ? 0.0f : LOG_EPS);
        lg[j2] = lv; mx = fmaxf(mx, lv);
      }
      #pragma unroll
      for (int off = 16; off > 0; off >>= 1) mx = fmaxf(mx, __shfl_xor(mx, off, 32));
      float se = 0.0f;
      #pragma unroll
      for (int j2 = 0; j2 < 4; ++j2) se += expf(lg[j2] - mx);
      #pragma unroll
      for (int off = 16; off > 0; off >>= 1) se += __shfl_xor(se, off, 32);
      float lse = logf(se);
      float bestv = -3.4e38f; int besti = 0;
      #pragma unroll
      for (int j2 = 0; j2 < 4; ++j2){
        int s = lane + 32*j2;
        bool msk = rowv && (s < L);
        float ls = lg[j2] - mx - lse;
        out_ls[((size_t)b*NS + t)*NS + s] = ls;
        float rep = msk ? ls : -1e7f;
        if (rep > bestv){ bestv = rep; besti = s; }
      }
      #pragma unroll
      for (int off = 16; off > 0; off >>= 1){
        float ov = __shfl_xor(bestv, off, 32);
        int   oi = __shfl_xor(besti, off, 32);
        if (ov > bestv || (ov == bestv && oi < besti)){ bestv = ov; besti = oi; }
      }
      if (lane == 0){ amS[b] = besti; out_am[b*NS + t] = (float)besti; }
    }
    __syncthreads();
    // ---- 6) dec_in = enc[b, argmax]  (A k-range [256,512)) ----
    for (int i = tid; i < NB*NH; i += NT_LSTM){
      int b = i >> 8, k = i & 255;
      Abuf[a_swz(b, 256 + k, 16)] = encS[((size_t)(b*NS + amS[b]))*NH + k];
    }
    __syncthreads();
  }
}

// ============================ host launcher ============================
extern "C" void kernel_launch(void* const* d_in, const int* in_sizes, int n_in,
                              void* d_out, int out_size, void* d_ws, size_t ws_size,
                              hipStream_t stream){
  (void)in_sizes; (void)n_in; (void)out_size; (void)ws_size;
  const float* input_seq = (const float*)d_in[0];
  const int*   lengths   = (const int*)  d_in[1];
  const float* W_emb = (const float*)d_in[2];
  const float* Wih_f = (const float*)d_in[3];
  const float* Whh_f = (const float*)d_in[4];
  const float* bih_f = (const float*)d_in[5];
  const float* bhh_f = (const float*)d_in[6];
  const float* Wih_b = (const float*)d_in[7];
  const float* Whh_b = (const float*)d_in[8];
  const float* bih_b = (const float*)d_in[9];
  const float* bhh_b = (const float*)d_in[10];
  const float* dWih  = (const float*)d_in[11];
  const float* dWhh  = (const float*)d_in[12];
  const float* dbih  = (const float*)d_in[13];
  const float* dbhh  = (const float*)d_in[14];
  const float* W1    = (const float*)d_in[15];
  const float* W2    = (const float*)d_in[16];
  const float* Vv    = (const float*)d_in[17];

  // workspace carve (~26 MB)
  char* ws = (char*)d_ws;
  size_t off = 0;
  auto carve = [&](size_t bytes)->char*{ char* p = ws + off; off = (off + bytes + 255) & ~(size_t)255; return p; };
  _Float16* emb   = (_Float16*)carve((size_t)NB*NS*NE*2);
  _Float16* WcatF = (_Float16*)carve((size_t)NG*384*2);
  _Float16* WcatB = (_Float16*)carve((size_t)NG*384*2);
  _Float16* WcatD = (_Float16*)carve((size_t)NG*512*2);
  _Float16* W1s   = (_Float16*)carve((size_t)NH*NH*2);
  _Float16* W2s   = (_Float16*)carve((size_t)NH*NH*2);
  float* biasF = (float*)carve(NG*4);
  float* biasB = (float*)carve(NG*4);
  float* biasD = (float*)carve(NG*4);
  float* hT    = (float*)carve((size_t)NB*NH*4);
  float* cT    = (float*)carve((size_t)NB*NH*4);
  _Float16* encF = (_Float16*)carve((size_t)NB*NS*NH*2);
  _Float16* encB = (_Float16*)carve((size_t)NB*NS*NH*2);
  _Float16* encS = (_Float16*)carve((size_t)NB*NS*NH*2);
  float* encT = (float*)carve((size_t)NB*NS*NH*4);

  float* out_ls   = (float*)d_out;                       // (B,S,S)
  float* out_am   = out_ls + (size_t)NB*NS*NS;           // (B,S)
  float* out_mask = out_am + (size_t)NB*NS;              // (B,S,S)

  // ---- prep ----
  pn_embed<<<(NB*NS*NE)/256, 256, 0, stream>>>(input_seq, W_emb, emb);
  pn_swzB<<<1536, 256, 0, stream>>>(WcatF, Whh_f, Wih_f, NG, NH, NE);
  pn_swzB<<<1536, 256, 0, stream>>>(WcatB, Whh_b, Wih_b, NG, NH, NE);
  pn_swzB<<<2048, 256, 0, stream>>>(WcatD, dWhh, dWih, NG, NH, NH);
  pn_swzB<<<256,  256, 0, stream>>>(W1s, W1, nullptr, NH, NH, 0);
  pn_swzB<<<256,  256, 0, stream>>>(W2s, W2, nullptr, NH, NH, 0);
  pn_bias<<<4, 256, 0, stream>>>(biasF, bih_f, bhh_f, NG);
  pn_bias<<<4, 256, 0, stream>>>(biasB, bih_b, bhh_b, NG);
  pn_bias<<<4, 256, 0, stream>>>(biasD, dbih, dbhh, NG);

  // ---- bidirectional LSTM (fwd + bwd concurrently) ----
  pn_bilstm<<<2, NT_LSTM, 53504, stream>>>(emb, lengths, WcatF, WcatB, biasF, biasB,
                                           encF, encB, hT, cT);

  // ---- enc combine, enc_trans GEMM, mask output ----
  pn_enc_combine<<<(NB*NS*NH)/256, 256, 0, stream>>>(encF, encB, encS);
  pn_enc_trans<<<1024, 256, 0, stream>>>(encS, W1s, encT);
  pn_mask<<<(NB*NS*NS)/256, 256, 0, stream>>>(lengths, out_mask);

  // ---- pointer decoder ----
  pn_decoder<<<1, NT_LSTM, 169472, stream>>>(WcatD, biasD, W2s, hT, cT, encS, encT,
                                             Vv, lengths, out_ls, out_am);
}
